// MultiHeadAttention_2894807958079
// MI455X (gfx1250) — matrix operand order repaired and checked by reading, not compiled, not run on hardware
//
#include <hip/hip_runtime.h>
#include <cstdint>

typedef __attribute__((ext_vector_type(16))) _Float16 v16h;
typedef __attribute__((ext_vector_type(8)))  _Float16 v8h;
typedef __attribute__((ext_vector_type(8)))  float    v8f;
typedef __attribute__((ext_vector_type(4)))  int      v4i;

#ifndef __has_builtin
#define __has_builtin(x) 0
#endif

#if __has_builtin(__builtin_amdgcn_global_load_async_to_lds_b128) && \
    __has_builtin(__builtin_amdgcn_s_wait_asynccnt)
#define ATTN_ASYNC_STAGE 1
typedef __attribute__((address_space(1))) v4i v4i_as1;
typedef __attribute__((address_space(3))) v4i v4i_as3;
#define CPY16(l, g)                                                     \
  __builtin_amdgcn_global_load_async_to_lds_b128((v4i_as1*)(g),         \
                                                 (v4i_as3*)(l), 0, 0)
#define WAIT_ASYNC() __builtin_amdgcn_s_wait_asynccnt(0)
#else
#define ATTN_ASYNC_STAGE 0
#define CPY16(l, g) (*(float4*)(l) = *(const float4*)(g))
#define WAIT_ASYNC() ((void)0)
#endif

static constexpr int Bb = 8, Ss = 2048, Hh = 8, Dd = 64;
static constexpr int KB = 64;        // keys per attention iteration
static constexpr int NIT = Ss / KB;  // 32 iterations

#define DEV static __device__ __forceinline__

DEV v8f wmma16(v16h a, v16h b, v8f c) {
  // D = A(16x32 f16) * B(32x16 f16) + C(16x16 f32)
  return __builtin_amdgcn_wmma_f32_16x16x32_f16(false, a, false, b,
                                                (short)0, c, false, false);
}

// ---- A-operand loaders (16x32, M x K). Row m = ln; per-lane chunks at
// k = (lh*8 .. +8) and k = 16+(lh*8) .. +8, relative to k0. ----
DEV v16h loadA_f16(const _Float16* base, int rs, int ln, int lh, int k0) {
  const _Float16* p = base + (size_t)ln * rs + k0 + lh * 8;
  v8h c1 = *(const v8h*)(p);
  v8h c2 = *(const v8h*)(p + 16);
  v16h a;
#pragma unroll
  for (int i = 0; i < 8; ++i) { a[i] = c1[i]; a[i + 8] = c2[i]; }
  return a;
}

DEV v16h loadA_f32(const float* base, int rs, int ln, int lh, int k0) {
  const float* p = base + (size_t)ln * rs + k0 + lh * 8;
  v16h a;
#pragma unroll
  for (int i = 0; i < 8; ++i) {
    a[i]     = (_Float16)p[i];
    a[i + 8] = (_Float16)p[16 + i];
  }
  return a;
}

// ---- B-operand loaders (32x16, K x N). Lane holds column n = ln; chunks
// at k = (lh*8 .. +8) and k = 16+(lh*8) .. +8, relative to k0. `rowbase` points at the row
// of the matrix whose rows correspond to columns of B (i.e. B = rows^T). ----
DEV v16h loadB_f16(const _Float16* rowbase, int rs, int ln, int lh, int k0) {
  const _Float16* p = rowbase + (size_t)ln * rs + k0 + lh * 8;
  v8h c1 = *(const v8h*)(p);
  v8h c2 = *(const v8h*)(p + 16);
  v16h b;
#pragma unroll
  for (int i = 0; i < 8; ++i) { b[i] = c1[i]; b[i + 8] = c2[i]; }
  return b;
}

DEV v16h loadB_f32(const float* rowbase, int rs, int ln, int lh, int k0) {
  const float* p = rowbase + (size_t)ln * rs + k0 + lh * 8;
  v16h b;
#pragma unroll
  for (int i = 0; i < 8; ++i) {
    b[i]     = (_Float16)p[i];
    b[i + 8] = (_Float16)p[16 + i];
  }
  return b;
}

// =====================================================================
// Kernel 1: QKV projection.  x[B,S,H,D] f32 -> Q16,K16 [B,H,S,D] f16,
//           Vt [B,H,D,S] f16 (transposed for the PV B-operand).
// =====================================================================
__global__ __launch_bounds__(128) void k_qkv(
    const float* __restrict__ x,
    const float* __restrict__ Wq, const float* __restrict__ bq,
    const float* __restrict__ Wk, const float* __restrict__ bk,
    const float* __restrict__ Wv, const float* __restrict__ bv,
    _Float16* __restrict__ Q16, _Float16* __restrict__ K16,
    _Float16* __restrict__ Vt) {
  const int bh = blockIdx.y;
  const int b = bh / Hh, h = bh % Hh;
  const int wave = threadIdx.x >> 5;
  const int lane = threadIdx.x & 31, ln = lane & 15, lh = lane >> 4;
  const int s0 = blockIdx.x * 64 + wave * 16;

  const float* xrow = x + (((size_t)b * Ss + s0) * Hh + h) * Dd;
  const int xs = Hh * Dd;  // row stride between consecutive s for fixed (b,h)
  v16h ax0 = loadA_f32(xrow, xs, ln, lh, 0);
  v16h ax1 = loadA_f32(xrow, xs, ln, lh, 32);

  const float* Ws[3] = {Wq, Wk, Wv};
  const float* bs[3] = {bq, bk, bv};
#pragma unroll
  for (int m = 0; m < 3; ++m) {
#pragma unroll
    for (int et = 0; et < 4; ++et) {
      // B[k][n] = W[et*16+n][k] : W rows are contiguous in k(=d).
      const float* wb = Ws[m] + (size_t)(et * 16) * Dd;
      v16h b0 = loadB_f32(wb, Dd, ln, lh, 0);
      v16h b1 = loadB_f32(wb, Dd, ln, lh, 32);
      v8f c = 0.0f;
      c = wmma16(ax0, b0, c);
      c = wmma16(ax1, b1, c);
      const int e = et * 16 + ln;
      const float bias = bs[m][e];
#pragma unroll
      for (int r = 0; r < 8; ++r) {
        const int s = s0 + r + lh * 8;  // C layout: M = r + (lane/16)*8
        const _Float16 val = (_Float16)(c[r] + bias);
        if (m == 0)      Q16[((size_t)bh * Ss + s) * Dd + e] = val;
        else if (m == 1) K16[((size_t)bh * Ss + s) * Dd + e] = val;
        else             Vt[((size_t)bh * Dd + e) * Ss + s] = val;
      }
    }
  }
}

// =====================================================================
// Kernel 2: flash attention. 4 waves/WG, each wave owns a 16-query tile.
// 64-key blocks, double-buffered ASYNCcnt staging of K/V tiles in LDS.
// =====================================================================
__global__ __launch_bounds__(128) void k_attn(
    const _Float16* __restrict__ Q16, const _Float16* __restrict__ K16,
    const _Float16* __restrict__ Vt, _Float16* __restrict__ O16) {
  __shared__ __align__(32) _Float16 sK[2][KB * Dd];   // rows = key, cols = d
  __shared__ __align__(32) _Float16 sV[2][Dd * KB];   // rows = d,  cols = key
  __shared__ __align__(32) _Float16 sP[4][16 * KB];   // per-wave P transpose

  const int bh = blockIdx.y;
  const int wave = threadIdx.x >> 5;
  const int lane = threadIdx.x & 31, ln = lane & 15, lh = lane >> 4;
  const int q0 = blockIdx.x * 64 + wave * 16;

  const _Float16* Qb = Q16 + ((size_t)bh * Ss + q0) * Dd;
  v16h aq0 = loadA_f16(Qb, Dd, ln, lh, 0);
  v16h aq1 = loadA_f16(Qb, Dd, ln, lh, 32);

  v8f acc[4];
  float mr[8], lr[8];
#pragma unroll
  for (int dt = 0; dt < 4; ++dt) acc[dt] = 0.0f;
#pragma unroll
  for (int r = 0; r < 8; ++r) { mr[r] = -3.0e38f; lr[r] = 0.0f; }

  const _Float16* Kb = K16 + (size_t)bh * Ss * Dd;
  const _Float16* Vb = Vt + (size_t)bh * Dd * Ss;
  _Float16* Pw = sP[wave];

  // ---- staging: K tile is 64*64 halves contiguous (8KB); V tile is 64
  // rows of 64 halves with row stride Ss. 512 x 16B chunks each, 128 thr. ----
#define STAGE_TILE(buf, kb)                                                  \
  do {                                                                       \
    _Float16* dK = sK[buf];                                                  \
    _Float16* dV = sV[buf];                                                  \
    const _Float16* srcK = Kb + (size_t)(kb) * Dd;                           \
    for (int i = threadIdx.x; i < 512; i += 128)                             \
      CPY16(dK + i * 8, srcK + i * 8);                                       \
    for (int i = threadIdx.x; i < 512; i += 128) {                           \
      const int d_ = i >> 3, off_ = (i & 7) * 8;                             \
      CPY16(dV + d_ * KB + off_, Vb + (size_t)d_ * Ss + (kb) + off_);        \
    }                                                                        \
  } while (0)

  STAGE_TILE(0, 0);
  WAIT_ASYNC();
  __syncthreads();

  for (int it = 0; it < NIT; ++it) {
    const int cur = it & 1;
    if (it + 1 < NIT) STAGE_TILE(1 - cur, (it + 1) * KB);  // prefetch overlap

    const _Float16* cK = sK[cur];
    const _Float16* cV = sV[cur];

    // ---- scores: four 16x16 tiles over 64 keys, K=64 in two WMMAs each ----
    v8f cs[4];
#pragma unroll
    for (int t = 0; t < 4; ++t) {
      const _Float16* kr = cK + (size_t)(t * 16) * Dd;  // B cols = key rows
      v16h b0 = loadB_f16(kr, Dd, ln, lh, 0);
      v16h b1 = loadB_f16(kr, Dd, ln, lh, 32);
      v8f c = 0.0f;
      c = wmma16(aq0, b0, c);
      c = wmma16(aq1, b1, c);
      cs[t] = c;
    }

    // ---- online softmax (rows live across 16-lane halves per C vgpr) ----
#pragma unroll
    for (int r = 0; r < 8; ++r) {
      float sv[4], bm;
#pragma unroll
      for (int t = 0; t < 4; ++t) sv[t] = cs[t][r] * 0.125f;  // 1/sqrt(64)
      bm = fmaxf(fmaxf(sv[0], sv[1]), fmaxf(sv[2], sv[3]));
      bm = fmaxf(bm, __shfl_xor(bm, 1, 32));
      bm = fmaxf(bm, __shfl_xor(bm, 2, 32));
      bm = fmaxf(bm, __shfl_xor(bm, 4, 32));
      bm = fmaxf(bm, __shfl_xor(bm, 8, 32));
      const float mnew = fmaxf(mr[r], bm);
      const float alpha = __expf(mr[r] - mnew);
      float p[4], rs = 0.0f;
#pragma unroll
      for (int t = 0; t < 4; ++t) { p[t] = __expf(sv[t] - mnew); rs += p[t]; }
      rs += __shfl_xor(rs, 1, 32);
      rs += __shfl_xor(rs, 2, 32);
      rs += __shfl_xor(rs, 4, 32);
      rs += __shfl_xor(rs, 8, 32);
      lr[r] = lr[r] * alpha + rs;
      mr[r] = mnew;
#pragma unroll
      for (int dt = 0; dt < 4; ++dt) acc[dt][r] *= alpha;
      const int mrow = r + lh * 8;  // C-layout row for this lane/vgpr
#pragma unroll
      for (int t = 0; t < 4; ++t)
        Pw[mrow * KB + t * 16 + ln] = (_Float16)p[t];
    }
    asm volatile("s_wait_dscnt 0" ::: "memory");

    // ---- P(16x64) * V(64x64): A from LDS scratch, B from transposed V ----
    v16h ap0 = loadA_f16(Pw, KB, ln, lh, 0);
    v16h ap1 = loadA_f16(Pw, KB, ln, lh, 32);
#pragma unroll
    for (int dt = 0; dt < 4; ++dt) {
      const _Float16* vr = cV + (size_t)(dt * 16) * KB;  // B cols = d rows
      v16h bv0 = loadB_f16(vr, KB, ln, lh, 0);
      v16h bv1 = loadB_f16(vr, KB, ln, lh, 32);
      acc[dt] = wmma16(ap0, bv0, acc[dt]);
      acc[dt] = wmma16(ap1, bv1, acc[dt]);
    }

    WAIT_ASYNC();      // prefetched tile landed in LDS
    __syncthreads();   // all waves done reading `cur`, staging visible
  }

  // ---- finalize: O /= rowsum, store f16 [B,H,S,D] ----
#pragma unroll
  for (int r = 0; r < 8; ++r) {
    const float inv = 1.0f / lr[r];
    const int s = q0 + r + lh * 8;
#pragma unroll
    for (int dt = 0; dt < 4; ++dt) {
      O16[((size_t)bh * Ss + s) * Dd + dt * 16 + ln] =
          (_Float16)(acc[dt][r] * inv);
    }
  }
}

// =====================================================================
// Kernel 3: output projection. O16 [B,H,S,D] f16 -> out [B,S,H,D] f32.
// =====================================================================
__global__ __launch_bounds__(128) void k_oproj(
    const _Float16* __restrict__ O16, const float* __restrict__ Wo,
    const float* __restrict__ bo, float* __restrict__ out) {
  const int bh = blockIdx.y;
  const int b = bh / Hh, h = bh % Hh;
  const int wave = threadIdx.x >> 5;
  const int lane = threadIdx.x & 31, ln = lane & 15, lh = lane >> 4;
  const int s0 = blockIdx.x * 64 + wave * 16;

  const _Float16* Ob = O16 + ((size_t)bh * Ss + s0) * Dd;
  v16h a0 = loadA_f16(Ob, Dd, ln, lh, 0);
  v16h a1 = loadA_f16(Ob, Dd, ln, lh, 32);

#pragma unroll
  for (int et = 0; et < 4; ++et) {
    const float* wb = Wo + (size_t)(et * 16) * Dd;
    v16h b0 = loadB_f32(wb, Dd, ln, lh, 0);
    v16h b1 = loadB_f32(wb, Dd, ln, lh, 32);
    v8f c = 0.0f;
    c = wmma16(a0, b0, c);
    c = wmma16(a1, b1, c);
    const int e = et * 16 + ln;
    const float bias = bo[e];
#pragma unroll
    for (int r = 0; r < 8; ++r) {
      const int s = s0 + r + lh * 8;
      out[(((size_t)b * Ss + s) * Hh + h) * Dd + e] = c[r] + bias;
    }
  }
}

extern "C" void kernel_launch(void* const* d_in, const int* in_sizes, int n_in,
                              void* d_out, int out_size, void* d_ws,
                              size_t ws_size, hipStream_t stream) {
  (void)in_sizes; (void)n_in; (void)out_size; (void)ws_size;
  const float* x  = (const float*)d_in[0];
  const float* Wq = (const float*)d_in[1];
  const float* bq = (const float*)d_in[2];
  const float* Wk = (const float*)d_in[3];
  const float* bk = (const float*)d_in[4];
  const float* Wv = (const float*)d_in[5];
  const float* bv = (const float*)d_in[6];
  const float* Wo = (const float*)d_in[7];
  const float* bo = (const float*)d_in[8];
  float* out = (float*)d_out;

  const size_t N = (size_t)Bb * Hh * Ss * Dd;  // 8M elements
  _Float16* Q16 = (_Float16*)d_ws;             // 16 MB each, 64 MB total
  _Float16* K16 = Q16 + N;
  _Float16* Vt  = K16 + N;
  _Float16* O16 = Vt + N;

  dim3 grid(Ss / 64, Bb * Hh);  // 32 x 64 blocks, 128 threads (4 wave32)
  k_qkv<<<grid, 128, 0, stream>>>(x, Wq, bq, Wk, bk, Wv, bv, Q16, K16, Vt);
  k_attn<<<grid, 128, 0, stream>>>(Q16, K16, Vt, O16);
  k_oproj<<<grid, 128, 0, stream>>>(O16, Wo, bo, out);
}
